// RecursiveBeliefNetwork_27779848470681
// MI455X (gfx1250) — matrix operand order repaired and checked by reading, compiled, not run on hardware
//
#include <hip/hip_runtime.h>
#include <math.h>

// ---------------- problem constants ----------------
#define BB 256
#define TT 64
#define VV 768
#define HH 768
#define LL 32
#define SDD 12
#define NHH 12
#define HDD 64
#define SS 63              // T - 1
#define SCALE_ATTN 0.125f  // 1/sqrt(64)
#define INV_TEMP 10.0f

typedef __attribute__((ext_vector_type(16))) __bf16 v16bf;
typedef __attribute__((ext_vector_type(8)))  float  v8f;
typedef __attribute__((ext_vector_type(4)))  int    v4i;

#if __has_builtin(__builtin_amdgcn_global_load_async_to_lds_b128) && \
    __has_builtin(__builtin_amdgcn_s_wait_asynccnt)
#define USE_ASYNC_LDS 1
#else
#define USE_ASYNC_LDS 0
#endif

// ---------------- helpers ----------------
__device__ __forceinline__ unsigned int f2bf(float f) {
    unsigned int u = __float_as_uint(f);
    return (u + 0x7FFFu + ((u >> 16) & 1u)) >> 16;  // RNE
}
__device__ __forceinline__ unsigned int pack_bf16(float lo, float hi) {
    return f2bf(lo) | (f2bf(hi) << 16);
}

__device__ __forceinline__ float block_reduce_sum(float v, float* sd) {
    int tid = threadIdx.x;
    sd[tid] = v;
    __syncthreads();
    for (int s = blockDim.x >> 1; s > 0; s >>= 1) {
        if (tid < s) sd[tid] += sd[tid + s];
        __syncthreads();
    }
    float r = sd[0];
    __syncthreads();
    return r;
}

// fp32 -> bf16 weight pre-conversion (once per launch; weights stay in L2)
__global__ void f32_to_bf16_kernel(const float* __restrict__ in,
                                   unsigned short* __restrict__ out, int n) {
    int i = blockIdx.x * blockDim.x + threadIdx.x;
    int e = i * 2;
    if (e < n) {
        unsigned int p = pack_bf16(in[e], in[e + 1]);  // n always even
        *(unsigned int*)(out + e) = p;
    }
}

// ---------------- WMMA GEMM: C[m,n] = sum_k A[m,k] * Wbf[n,k] + bias[n] ----------------
// A: fp32 (converted to bf16 while staging). Wbf: pre-converted bf16
// (W tile staged with GLOBAL_LOAD_ASYNC_TO_LDS_B128 when available).
// REQUIRES: M % 64 == 0, K % 32 == 0, lda % 4 == 0, ldw % 8 == 0, bias != null.
// N arbitrary (clamped reads, guarded stores).
// Block tile 64(M) x 128(N); 8 waves; wave tile 32x32 (4 WMMAs per K-chunk).
__global__ __launch_bounds__(256) void gemm_bf16(
    const float* __restrict__ A, int lda,
    const unsigned short* __restrict__ Wbf, int ldw,
    const float* __restrict__ bias,
    float* __restrict__ C, int ldc,
    int M, int N, int K)
{
    // rows of 32 bf16 padded to 80 B (= 5*16) so every uint4 slot is 16B-aligned
    __shared__ unsigned int As[64][20];
    __shared__ unsigned int Ws[128][20];

    const int tid  = threadIdx.x;
    const int wave = tid >> 5;
    const int lane = tid & 31;
    const int half = lane >> 4;   // 0 or 1
    const int r    = lane & 15;

    const int m_blk = blockIdx.y * 64;
    const int n_blk = blockIdx.x * 128;
    const int wm = (wave >> 2) * 32;   // 0 / 32
    const int wn = (wave & 3) * 32;    // 0 / 32 / 64 / 96

    // A staging: float4 granularity (8 float4 per 32-wide row)
    const int arow0 = tid >> 3;   // rows 0..31 (i=0), 32..63 (i=1)
    const int ak4   = tid & 7;
    // W staging: uint4 granularity (4 uint4 of bf16 per 32-wide row)
    const int wrow0 = tid >> 2;   // rows 0..63 (i=0), 64..127 (i=1)
    const int wq4   = tid & 3;

    v8f acc[2][2] = {};

    for (int kk = 0; kk < K; kk += 32) {
        // ---- stage W tile 128x32: bf16 copy (async direct-to-LDS if avail) ----
        #pragma unroll
        for (int i = 0; i < 2; ++i) {
            int row = wrow0 + i * 64;
            int gn  = n_blk + row;
            int gnc = (gn < N) ? gn : (N - 1);  // branchless clamp (valid read)
            const unsigned short* gsrc = Wbf + (size_t)gnc * ldw + kk + wq4 * 8;
#if USE_ASYNC_LDS
            __builtin_amdgcn_global_load_async_to_lds_b128(
                (__attribute__((address_space(1))) v4i*)(uintptr_t)gsrc,
                (__attribute__((address_space(3))) v4i*)(uintptr_t)(&Ws[row][wq4 * 4]),
                0, 0);
#else
            const uint4 w = *(const uint4*)gsrc;
            *(uint4*)(&Ws[row][wq4 * 4]) = w;
#endif
        }
        // ---- stage A tile 64x32: 2 float4 per thread, convert to bf16 ----
        #pragma unroll
        for (int i = 0; i < 2; ++i) {
            int row = arow0 + i * 32;
            const float4 f = *(const float4*)(
                A + (size_t)(m_blk + row) * lda + kk + ak4 * 4);
            uint2 p;
            p.x = pack_bf16(f.x, f.y);
            p.y = pack_bf16(f.z, f.w);
            *(uint2*)(&As[row][ak4 * 2]) = p;
        }
#if USE_ASYNC_LDS
        __builtin_amdgcn_s_wait_asynccnt(0);  // wave's async W tile landed in LDS
#endif
        __syncthreads();

        union Frag { v16bf v; uint4 q[2]; };
        Frag a[2], b[2];
        // A fragment (16x32): dwords {4h..4h+3} and {8+4h..8+4h+3} => uint4 {h, h+2}
        #pragma unroll
        for (int mi = 0; mi < 2; ++mi) {
            const uint4* arow = (const uint4*)(&As[wm + mi * 16 + r][0]);
            a[mi].q[0] = arow[half];
            a[mi].q[1] = arow[half + 2];
        }
        // B fragment (32x16): dwords {4h..4h+3} and {8+4h..8+4h+3} => uint4 {h, h+2}
        #pragma unroll
        for (int ni = 0; ni < 2; ++ni) {
            const uint4* wrow = (const uint4*)(&Ws[wn + ni * 16 + r][0]);
            b[ni].q[0] = wrow[half];
            b[ni].q[1] = wrow[half + 2];
        }
        #pragma unroll
        for (int mi = 0; mi < 2; ++mi)
            #pragma unroll
            for (int ni = 0; ni < 2; ++ni)
                acc[mi][ni] = __builtin_amdgcn_wmma_f32_16x16x32_bf16(
                    false, a[mi].v, false, b[ni].v,
                    (short)0, acc[mi][ni], false, false);
        __syncthreads();
    }

    // store: D vgpr j -> row j + 8*half, col = r
    #pragma unroll
    for (int mi = 0; mi < 2; ++mi) {
        #pragma unroll
        for (int ni = 0; ni < 2; ++ni) {
            int n = n_blk + wn + ni * 16 + r;
            if (n >= N) continue;
            float bv = bias[n];
            int mbase = m_blk + wm + mi * 16 + 8 * half;
            #pragma unroll
            for (int j = 0; j < 8; ++j) {
                int mm = mbase + j;
                C[(size_t)mm * ldc + n] = acc[mi][ni][j] + bv;
            }
        }
    }
}

// ---------------- small kernels ----------------
__global__ void fill_kernel(float* p, int n, float v) {
    int i = blockIdx.x * blockDim.x + threadIdx.x;
    if (i < n) p[i] = v;
}

// x_t = state_seq[:,t] @ w_st^T + b_st  ->  tok[:, 1, :]
__global__ void xst_kernel(const float* __restrict__ state,
                           const float* __restrict__ w_st,
                           const float* __restrict__ b_st,
                           float* __restrict__ tok, int t) {
    int idx = blockIdx.x * blockDim.x + threadIdx.x;
    if (idx >= BB * HH) return;
    int b = idx / HH, h = idx % HH;
    const float* sp = state + (size_t)b * TT * SDD + (size_t)t * SDD;
    float acc = b_st[h];
    #pragma unroll
    for (int i = 0; i < SDD; ++i) acc += sp[i] * w_st[(size_t)h * SDD + i];
    tok[((size_t)b * 3 + 1) * HH + h] = acc;
}

// 3-token attention; only query token 2 is consumed downstream.
__global__ void attn_kernel(const float* __restrict__ q,
                            const float* __restrict__ k,
                            const float* __restrict__ v,
                            float* __restrict__ ctx2) {
    int idx = blockIdx.x * blockDim.x + threadIdx.x;
    if (idx >= BB * NHH) return;
    int b = idx / NHH, nh = idx % NHH;
    const float* qp = q + ((size_t)b * 3 + 2) * HH + nh * HDD;
    float s0 = 0.f, s1 = 0.f, s2 = 0.f;
    const float* k0 = k + ((size_t)b * 3 + 0) * HH + nh * HDD;
    const float* k1 = k + ((size_t)b * 3 + 1) * HH + nh * HDD;
    const float* k2 = k + ((size_t)b * 3 + 2) * HH + nh * HDD;
    for (int i = 0; i < HDD; ++i) {
        float qv = qp[i];
        s0 += qv * k0[i]; s1 += qv * k1[i]; s2 += qv * k2[i];
    }
    s0 *= SCALE_ATTN; s1 *= SCALE_ATTN; s2 *= SCALE_ATTN;
    float mx = fmaxf(s0, fmaxf(s1, s2));
    float e0 = expf(s0 - mx), e1 = expf(s1 - mx), e2 = expf(s2 - mx);
    float inv = 1.0f / (e0 + e1 + e2);
    float a0 = e0 * inv, a1 = e1 * inv, a2 = e2 * inv;
    const float* v0 = v + ((size_t)b * 3 + 0) * HH + nh * HDD;
    const float* v1 = v + ((size_t)b * 3 + 1) * HH + nh * HDD;
    const float* v2 = v + ((size_t)b * 3 + 2) * HH + nh * HDD;
    float* cp = ctx2 + (size_t)b * HH + nh * HDD;
    for (int i = 0; i < HDD; ++i)
        cp[i] = a0 * v0[i] + a1 * v1[i] + a2 * v2[i];
}

// C[row, :] = concat(A[row, :wA], Bsrc[row, :wB])  (per-input row strides)
__global__ void concat2_kernel(const float* __restrict__ A, int ldA, int wA,
                               const float* __restrict__ Bsrc, int ldB, int wB,
                               float* __restrict__ C, int rows) {
    int w = wA + wB;
    int idx = blockIdx.x * blockDim.x + threadIdx.x;
    if (idx >= rows * w) return;
    int row = idx / w, c = idx % w;
    float val = (c < wA) ? A[(size_t)row * ldA + c]
                         : Bsrc[(size_t)row * ldB + (c - wA)];
    C[(size_t)row * w + c] = val;
}

// z = mu_q + exp(ls_q)*eps ; kl per-batch sum over L (wave32 reduce)
__global__ void zkl_kernel(const float* __restrict__ pr,
                           const float* __restrict__ po,
                           const float* __restrict__ eps,
                           float* __restrict__ z,
                           float* __restrict__ kl_arr, int s) {
    int b = blockIdx.x, l = threadIdx.x;  // block = 32 threads
    float mu_p = pr[b * 64 + l],      ls_p = pr[b * 64 + 32 + l];
    float mu_q = po[b * 64 + l],      ls_q = po[b * 64 + 32 + l];
    float e = eps[(size_t)b * SS * LL + (size_t)s * LL + l];
    float sq = expf(ls_q), sp = expf(ls_p);
    z[b * LL + l] = mu_q + sq * e;
    float dmu = mu_q - mu_p;
    float kl = ls_p - ls_q + (sq * sq + dmu * dmu) / (2.0f * sp * sp) - 0.5f;
    for (int off = 16; off > 0; off >>= 1) kl += __shfl_xor(kl, off);
    if (l == 0) kl_arr[s * BB + b] = kl;
}

// GRU gate fusion; gi/gh already contain biases via GEMM epilogue.
// If bp != null, also emit b_t = relu(h') to bp and belief output slice s.
__global__ void gru_combine_kernel(const float* __restrict__ gi,
                                   const float* __restrict__ gh,
                                   float* __restrict__ h,
                                   float* __restrict__ bp,
                                   float* __restrict__ belief_out, int s) {
    int idx = blockIdx.x * blockDim.x + threadIdx.x;
    if (idx >= BB * HH) return;
    int b = idx / HH, hh = idx % HH;
    size_t base = (size_t)b * 3 * HH;
    float ir = gi[base + hh], iz = gi[base + HH + hh], in_ = gi[base + 2 * HH + hh];
    float hr = gh[base + hh], hz = gh[base + HH + hh], hn  = gh[base + 2 * HH + hh];
    float rg = 1.0f / (1.0f + expf(-(ir + hr)));
    float zg = 1.0f / (1.0f + expf(-(iz + hz)));
    float ng = tanhf(in_ + rg * hn);
    float hnew = (1.0f - zg) * ng + zg * h[idx];
    h[idx] = hnew;
    if (bp) {
        float bt = fmaxf(hnew, 0.0f);
        bp[idx] = bt;
        belief_out[((size_t)b * SS + s) * HH + hh] = bt;
    }
}

__global__ void gelu_kernel(float* __restrict__ p, int n) {
    int i = blockIdx.x * blockDim.x + threadIdx.x;
    if (i < n) {
        float x = p[i];
        p[i] = 0.5f * x * (1.0f + erff(x * 0.70710678118654752f));
    }
}

// in-place L2 normalization of feat rows (B*S rows of H)
__global__ void normalize_kernel(float* __restrict__ feat) {
    __shared__ float sd[256];
    int row = blockIdx.x;
    float* f = feat + (size_t)row * HH;
    float ss = 0.f;
    for (int h = threadIdx.x; h < HH; h += 256) ss += f[h] * f[h];
    float tot = block_reduce_sum(ss, sd);
    float scale = 1.0f / fmaxf(sqrtf(tot), 1e-12f);
    for (int h = threadIdx.x; h < HH; h += 256) f[h] *= scale;
}

// InfoNCE: for (b, s<S-1) compute -(logit[s+1] - logsumexp_t logit[t])
__global__ void nce_kernel(const float* __restrict__ fn,
                           float* __restrict__ nce_arr) {
    __shared__ float sd[64];
    int bs = blockIdx.x;
    int b = bs / (SS - 1), s = bs % (SS - 1);
    int t = threadIdx.x;  // 64 threads, t in [0,63); t==63 inactive
    float logit = -1e30f;
    if (t < SS) {
        const float* fs = fn + ((size_t)b * SS + s) * HH;
        const float* ft = fn + ((size_t)b * SS + t) * HH;
        float d = 0.f;
        for (int h = 0; h < HH; ++h) d += fs[h] * ft[h];
        logit = d * INV_TEMP;
    }
    sd[t] = logit;
    __syncthreads();
    for (int o = 32; o > 0; o >>= 1) {
        if (t < o) sd[t] = fmaxf(sd[t], sd[t + o]);
        __syncthreads();
    }
    float mx = sd[0];
    __syncthreads();
    float e = (t < SS) ? expf(logit - mx) : 0.0f;
    sd[t] = e;
    __syncthreads();
    for (int o = 32; o > 0; o >>= 1) {
        if (t < o) sd[t] += sd[t + o];
        __syncthreads();
    }
    float lse = mx + logf(sd[0]);
    if (t == s + 1) nce_arr[bs] = -(logit - lse);
}

// partial sums of (pred - target)^2 (deterministic grid/tree)
__global__ void recon_partial_kernel(const float* __restrict__ pred,
                                     const float* __restrict__ vis,
                                     float* __restrict__ out) {
    __shared__ float sd[256];
    const long n = (long)BB * SS * VV;
    float acc = 0.f;
    for (long e = (long)blockIdx.x * blockDim.x + threadIdx.x; e < n;
         e += (long)gridDim.x * blockDim.x) {
        long b = e / ((long)SS * VV);
        long rem = e % ((long)SS * VV);
        long s = rem / VV, v = rem % VV;
        float d = pred[e] - vis[b * (long)TT * VV + (s + 1) * VV + v];
        acc += d * d;
    }
    float tot = block_reduce_sum(acc, sd);
    if (threadIdx.x == 0) out[blockIdx.x] = tot;
}

__global__ void sum_partial_kernel(const float* __restrict__ in, int n,
                                   float* __restrict__ out) {
    __shared__ float sd[256];
    float acc = 0.f;
    for (int e = blockIdx.x * blockDim.x + threadIdx.x; e < n;
         e += gridDim.x * blockDim.x) acc += in[e];
    float tot = block_reduce_sum(acc, sd);
    if (threadIdx.x == 0) out[blockIdx.x] = tot;
}

__global__ void final_kernel(const float* __restrict__ rpart, int nr,
                             const float* __restrict__ kpart, int nk,
                             const float* __restrict__ npart, int nn,
                             float* __restrict__ out) {
    __shared__ float sd[256];
    float a = 0.f;
    for (int i = threadIdx.x; i < nr; i += 256) a += rpart[i];
    float rsum = block_reduce_sum(a, sd);
    a = 0.f;
    for (int i = threadIdx.x; i < nk; i += 256) a += kpart[i];
    float ksum = block_reduce_sum(a, sd);
    a = 0.f;
    for (int i = threadIdx.x; i < nn; i += 256) a += npart[i];
    float nsum = block_reduce_sum(a, sd);
    if (threadIdx.x == 0) {
        float recon = rsum / (float)((long)BB * SS * VV);
        float klm   = ksum / (float)(SS * BB);
        float nce   = nsum / (float)(BB * (SS - 1));
        out[0] = recon + klm + nce;
        out[1] = recon;
        out[2] = klm;
        out[3] = nce;
    }
}

// ---------------- host-side orchestration ----------------
static inline void launch_gemm(const float* A, int lda,
                               const unsigned short* Wbf, int ldw,
                               const float* bias, float* C, int ldc,
                               int M, int N, int K, hipStream_t st) {
    dim3 grid((N + 127) / 128, (M + 63) / 64);
    gemm_bf16<<<grid, 256, 0, st>>>(A, lda, Wbf, ldw, bias, C, ldc, M, N, K);
}

static inline void launch_cvt(const float* in, unsigned short* out, int n,
                              hipStream_t st) {
    int pairs = n / 2;
    f32_to_bf16_kernel<<<(pairs + 255) / 256, 256, 0, st>>>(in, out, n);
}

extern "C" void kernel_launch(void* const* d_in, const int* in_sizes, int n_in,
                              void* d_out, int out_size, void* d_ws, size_t ws_size,
                              hipStream_t stream) {
    const float* visual = (const float*)d_in[0];
    const float* state  = (const float*)d_in[1];
    const float* eps    = (const float*)d_in[2];
    const float* w_vis  = (const float*)d_in[3];
    const float* b_vis  = (const float*)d_in[4];
    const float* w_st   = (const float*)d_in[5];
    const float* b_st   = (const float*)d_in[6];
    const float* w_bt   = (const float*)d_in[7];
    const float* b_bt   = (const float*)d_in[8];
    const float* w_q    = (const float*)d_in[9];
    const float* b_q    = (const float*)d_in[10];
    const float* w_k    = (const float*)d_in[11];
    const float* b_k    = (const float*)d_in[12];
    const float* w_v    = (const float*)d_in[13];
    const float* b_v    = (const float*)d_in[14];
    const float* w_o    = (const float*)d_in[15];
    const float* b_o    = (const float*)d_in[16];
    const float* w_pr   = (const float*)d_in[17];
    const float* b_pr   = (const float*)d_in[18];
    const float* w_po   = (const float*)d_in[19];
    const float* b_po   = (const float*)d_in[20];
    const float* w_ih0  = (const float*)d_in[21];
    const float* w_hh0  = (const float*)d_in[22];
    const float* b_ih0  = (const float*)d_in[23];
    const float* b_hh0  = (const float*)d_in[24];
    const float* w_ih1  = (const float*)d_in[25];
    const float* w_hh1  = (const float*)d_in[26];
    const float* b_ih1  = (const float*)d_in[27];
    const float* b_hh1  = (const float*)d_in[28];
    const float* w_d1   = (const float*)d_in[29];
    const float* b_d1   = (const float*)d_in[30];
    const float* w_d2   = (const float*)d_in[31];
    const float* b_d2   = (const float*)d_in[32];

    float* out   = (float*)d_out;
    float* outP  = out + 4;                       // pred_next  (B,S,V)
    float* outBl = outP + (size_t)BB * SS * VV;   // belief_traj (B,S,H)

    // workspace layout (float granularity; bf16 regions sized elems/2 floats)
    float* ws = (float*)d_ws;
    size_t o = 0;
    float* bp      = ws + o; o += (size_t)BB * HH;
    float* h0      = ws + o; o += (size_t)BB * HH;
    float* h1      = ws + o; o += (size_t)BB * HH;
    float* tok     = ws + o; o += (size_t)BB * 3 * HH;   // (B,3,H) == (3B,H)
    float* qb      = ws + o; o += (size_t)BB * 3 * HH;
    float* kb      = ws + o; o += (size_t)BB * 3 * HH;
    float* vb      = ws + o; o += (size_t)BB * 3 * HH;
    float* ctx2    = ws + o; o += (size_t)BB * HH;
    float* prb     = ws + o; o += (size_t)BB * 64;
    float* pob     = ws + o; o += (size_t)BB * 64;
    float* zb      = ws + o; o += (size_t)BB * LL;
    float* cat800  = ws + o; o += (size_t)BB * (HH + LL);
    float* cat1536 = ws + o; o += (size_t)BB * 2 * HH;
    float* gi      = ws + o; o += (size_t)BB * 3 * HH;
    float* gh      = ws + o; o += (size_t)BB * 3 * HH;
    float* hdec    = ws + o; o += (size_t)BB * HH;
    float* feat    = ws + o; o += (size_t)BB * SS * HH;  // (B,S,H)
    float* klarr   = ws + o; o += (size_t)SS * BB;
    float* ncearr  = ws + o; o += (size_t)BB * (SS - 1);
    float* rpart   = ws + o; o += 256;
    float* kpart   = ws + o; o += 32;
    float* npart   = ws + o; o += 32;
    // bf16 weight mirrors (element counts are all even & multiples of 8)
    unsigned short* wb_vis = (unsigned short*)(ws + o); o += (size_t)HH * VV / 2;
    unsigned short* wb_bt  = (unsigned short*)(ws + o); o += (size_t)HH * HH / 2;
    unsigned short* wb_q   = (unsigned short*)(ws + o); o += (size_t)HH * HH / 2;
    unsigned short* wb_k   = (unsigned short*)(ws + o); o += (size_t)HH * HH / 2;
    unsigned short* wb_v   = (unsigned short*)(ws + o); o += (size_t)HH * HH / 2;
    unsigned short* wb_o   = (unsigned short*)(ws + o); o += (size_t)HH * HH / 2;
    unsigned short* wb_pr  = (unsigned short*)(ws + o); o += (size_t)64 * HH / 2;
    unsigned short* wb_po  = (unsigned short*)(ws + o); o += (size_t)64 * 2 * HH / 2;
    unsigned short* wb_ih0 = (unsigned short*)(ws + o); o += (size_t)3 * HH * (HH + LL) / 2;
    unsigned short* wb_hh0 = (unsigned short*)(ws + o); o += (size_t)3 * HH * HH / 2;
    unsigned short* wb_ih1 = (unsigned short*)(ws + o); o += (size_t)3 * HH * HH / 2;
    unsigned short* wb_hh1 = (unsigned short*)(ws + o); o += (size_t)3 * HH * HH / 2;
    unsigned short* wb_d1  = (unsigned short*)(ws + o); o += (size_t)HH * (HH + LL) / 2;
    unsigned short* wb_d2  = (unsigned short*)(ws + o); o += (size_t)VV * HH / 2;
    (void)ws_size; (void)in_sizes; (void)n_in; (void)out_size;

    // ---- one-time (per launch) weight conversion ----
    launch_cvt(w_vis, wb_vis, HH * VV, stream);
    launch_cvt(w_bt,  wb_bt,  HH * HH, stream);
    launch_cvt(w_q,   wb_q,   HH * HH, stream);
    launch_cvt(w_k,   wb_k,   HH * HH, stream);
    launch_cvt(w_v,   wb_v,   HH * HH, stream);
    launch_cvt(w_o,   wb_o,   HH * HH, stream);
    launch_cvt(w_pr,  wb_pr,  64 * HH, stream);
    launch_cvt(w_po,  wb_po,  64 * 2 * HH, stream);
    launch_cvt(w_ih0, wb_ih0, 3 * HH * (HH + LL), stream);
    launch_cvt(w_hh0, wb_hh0, 3 * HH * HH, stream);
    launch_cvt(w_ih1, wb_ih1, 3 * HH * HH, stream);
    launch_cvt(w_hh1, wb_hh1, 3 * HH * HH, stream);
    launch_cvt(w_d1,  wb_d1,  HH * (HH + LL), stream);
    launch_cvt(w_d2,  wb_d2,  VV * HH, stream);

    const int BH = BB * HH;
    fill_kernel<<<(BH + 255) / 256, 256, 0, stream>>>(bp, BH, 0.0f);
    fill_kernel<<<(BH + 255) / 256, 256, 0, stream>>>(h0, BH, 0.0f);
    fill_kernel<<<(BH + 255) / 256, 256, 0, stream>>>(h1, BH, 0.0f);

    for (int s = 0; s < SS; ++s) {
        // tok[:,2] = visual[:,s] @ w_vis^T + b_vis
        launch_gemm(visual + (size_t)s * VV, TT * VV, wb_vis, VV, b_vis,
                    tok + 2 * HH, 3 * HH, BB, HH, VV, stream);
        // tok[:,1] = state[:,s] @ w_st^T + b_st   (K=12, plain VALU)
        xst_kernel<<<(BH + 255) / 256, 256, 0, stream>>>(state, w_st, b_st, tok, s);
        // tok[:,0] = bp @ w_bt^T + b_bt
        launch_gemm(bp, HH, wb_bt, HH, b_bt, tok, 3 * HH, BB, HH, HH, stream);

        // q, k, v projections on stacked tokens (3B rows)
        launch_gemm(tok, HH, wb_q, HH, b_q, qb, HH, 3 * BB, HH, HH, stream);
        launch_gemm(tok, HH, wb_k, HH, b_k, kb, HH, 3 * BB, HH, HH, stream);
        launch_gemm(tok, HH, wb_v, HH, b_v, vb, HH, 3 * BB, HH, HH, stream);

        attn_kernel<<<(BB * NHH + 255) / 256, 256, 0, stream>>>(qb, kb, vb, ctx2);

        // f_t = ctx2 @ w_o^T + b_o  -> written straight into feat[b, s, :]
        launch_gemm(ctx2, HH, wb_o, HH, b_o, feat + (size_t)s * HH, SS * HH,
                    BB, HH, HH, stream);

        // prior / posterior heads
        launch_gemm(bp, HH, wb_pr, HH, b_pr, prb, 64, BB, 64, HH, stream);
        concat2_kernel<<<(BB * 2 * HH + 255) / 256, 256, 0, stream>>>(
            bp, HH, HH, feat + (size_t)s * HH, SS * HH, HH, cat1536, BB);
        launch_gemm(cat1536, 2 * HH, wb_po, 2 * HH, b_po, pob, 64, BB, 64, 2 * HH, stream);

        zkl_kernel<<<BB, 32, 0, stream>>>(prb, pob, eps, zb, klarr, s);

        // GRU0: input = [f_t, z]
        concat2_kernel<<<(BB * (HH + LL) + 255) / 256, 256, 0, stream>>>(
            feat + (size_t)s * HH, SS * HH, HH, zb, LL, LL, cat800, BB);
        launch_gemm(cat800, HH + LL, wb_ih0, HH + LL, b_ih0, gi, 3 * HH, BB, 3 * HH, HH + LL, stream);
        launch_gemm(h0, HH, wb_hh0, HH, b_hh0, gh, 3 * HH, BB, 3 * HH, HH, stream);
        gru_combine_kernel<<<(BH + 255) / 256, 256, 0, stream>>>(gi, gh, h0, nullptr, nullptr, s);

        // GRU1: input = h0'
        launch_gemm(h0, HH, wb_ih1, HH, b_ih1, gi, 3 * HH, BB, 3 * HH, HH, stream);
        launch_gemm(h1, HH, wb_hh1, HH, b_hh1, gh, 3 * HH, BB, 3 * HH, HH, stream);
        gru_combine_kernel<<<(BH + 255) / 256, 256, 0, stream>>>(gi, gh, h1, bp, outBl, s);

        // decoder
        concat2_kernel<<<(BB * (HH + LL) + 255) / 256, 256, 0, stream>>>(
            bp, HH, HH, zb, LL, LL, cat800, BB);
        launch_gemm(cat800, HH + LL, wb_d1, HH + LL, b_d1, hdec, HH, BB, HH, HH + LL, stream);
        gelu_kernel<<<(BH + 255) / 256, 256, 0, stream>>>(hdec, BH);
        launch_gemm(hdec, HH, wb_d2, HH, b_d2, outP + (size_t)s * VV, SS * VV,
                    BB, VV, HH, stream);
    }

    // losses
    recon_partial_kernel<<<256, 256, 0, stream>>>(outP, visual, rpart);
    normalize_kernel<<<BB * SS, 256, 0, stream>>>(feat);
    nce_kernel<<<BB * (SS - 1), 64, 0, stream>>>(feat, ncearr);
    sum_partial_kernel<<<32, 256, 0, stream>>>(klarr, SS * BB, kpart);
    sum_partial_kernel<<<32, 256, 0, stream>>>(ncearr, BB * (SS - 1), npart);
    final_kernel<<<1, 256, 0, stream>>>(rpart, 256, kpart, 32, npart, 32, out);
}
